// GNNPredictor_58368605553172
// MI455X (gfx1250) — matrix operand order repaired and checked by reading, not compiled, not run on hardware
//
#include <hip/hip_runtime.h>
#include <hip/hip_bf16.h>

#define N_NODES 200000
#define N_EDGES 800000
#define N_GRAPHS 4000
#define D_N 35
#define IN_CH 8
#define EMB 256
#define HID 128

typedef __attribute__((ext_vector_type(16))) __bf16 v16bf;
typedef __attribute__((ext_vector_type(8)))  __bf16 v8bf;
typedef __attribute__((ext_vector_type(2)))  __bf16 bf16x2;
typedef __attribute__((ext_vector_type(8)))  float  v8f;

// LDS A-panel stride (elements): 256 + 8 pad -> rows hit distinct bank groups
#define A_STRIDE 264

__device__ __forceinline__ unsigned short bfbits(float f) {
    unsigned u = __builtin_bit_cast(unsigned, f);
    u += 0x7fffu + ((u >> 16) & 1u);            // round-to-nearest-even
    return (unsigned short)(u >> 16);
}
__device__ __forceinline__ __bf16 f2bf(float f) {
#if __has_builtin(__builtin_amdgcn_cvt_pk_bf16_f32)
    bf16x2 p = __builtin_amdgcn_cvt_pk_bf16_f32(f, f);
    return p[0];
#else
    return __builtin_bit_cast(__bf16, bfbits(f));
#endif
}
__device__ __forceinline__ float bf2f(unsigned short u) {
    return __builtin_bit_cast(float, (unsigned)u << 16);  // exact widening
}

// ---------------------------------------------------------------------------
__global__ void zero_kernel(float* __restrict__ p, size_t n) {
    size_t i = (size_t)blockIdx.x * blockDim.x + threadIdx.x;
    size_t stride = (size_t)gridDim.x * blockDim.x;
    for (; i < n; i += stride) p[i] = 0.0f;
}

// One-time weight conversion f32 -> bf16 (W is 256x256; stays L2-resident)
__global__ void __launch_bounds__(256)
cvtw_kernel(const float* __restrict__ W, unsigned short* __restrict__ Wbf, int n) {
    int i = blockIdx.x * 256 + threadIdx.x;
    if (i < n) Wbf[i] = bfbits(W[i]);
}

// ---------------------------------------------------------------------------
// h0[n, 0:8] = x[n, 0:35] @ W_emb.T      (thread = (node, out-channel))
// ---------------------------------------------------------------------------
__global__ void __launch_bounds__(256)
emb_kernel(const float* __restrict__ x, const float* __restrict__ We,
           float* __restrict__ h0) {
    __shared__ float sW[IN_CH * D_N];
    for (int i = threadIdx.x; i < IN_CH * D_N; i += 256) sW[i] = We[i];
    __syncthreads();
    int t = blockIdx.x * 256 + threadIdx.x;
    int node = t >> 3, c = t & 7;
    if (node >= N_NODES) return;
    const float* xr = x + (size_t)node * D_N;
    float acc = 0.0f;
#pragma unroll
    for (int k = 0; k < D_N; ++k) acc += xr[k] * sW[c * D_N + k];
    h0[(size_t)node * IN_CH + c] = acc;
}

// ---------------------------------------------------------------------------
// Edge scatter-sum, 8-wide features: m8[dst] += h0[src]   (f32 atomics)
// ---------------------------------------------------------------------------
__global__ void __launch_bounds__(256)
scatter8_kernel(const float* __restrict__ h0, const int* __restrict__ src,
                const int* __restrict__ dst, float* __restrict__ m8) {
    int e = blockIdx.x * 256 + threadIdx.x;
    if (e >= N_EDGES) return;
    int s = src[e], d = dst[e];
#pragma unroll
    for (int c = 0; c < IN_CH; ++c)
        atomicAdd(&m8[(size_t)d * IN_CH + c], h0[(size_t)s * IN_CH + c]);
}

// ---------------------------------------------------------------------------
// Edge scatter-sum, 256-wide: gather bf16 activations (half the read bytes),
// widen exactly to f32, accumulate with f32 global atomics.
// Block = edge, thread = feature (coalesced).
// ---------------------------------------------------------------------------
__global__ void __launch_bounds__(256)
scatter256_kernel(const unsigned short* __restrict__ hb, const int* __restrict__ src,
                  const int* __restrict__ dst, float* __restrict__ m) {
    int e = blockIdx.x;
    int s = src[e], d = dst[e];
    int c = threadIdx.x;
    float v = bf2f(hb[(size_t)s * EMB + c]);
    atomicAdd(&m[(size_t)d * EMB + c], v);
}

// ---------------------------------------------------------------------------
// Layer 1 (K=8, tiny): h1 = relu(m8 @ W1.T + b1); also emits bf16 shadow copy
// ---------------------------------------------------------------------------
__global__ void __launch_bounds__(256)
layer1_kernel(const float* __restrict__ m8, const float* __restrict__ W1,
              const float* __restrict__ b1, float* __restrict__ out,
              unsigned short* __restrict__ outbf) {
    __shared__ float sW[EMB * IN_CH];
    for (int i = threadIdx.x; i < EMB * IN_CH; i += 256) sW[i] = W1[i];
    __syncthreads();
    int o = threadIdx.x;
    int base = blockIdx.x * 16;
    float bo = b1[o];
    for (int n = 0; n < 16; ++n) {
        int node = base + n;
        float acc = bo;
#pragma unroll
        for (int k = 0; k < IN_CH; ++k)
            acc += m8[(size_t)node * IN_CH + k] * sW[o * IN_CH + k];
        acc = acc > 0.0f ? acc : 0.0f;
        out[(size_t)node * EMB + o] = acc;
        outbf[(size_t)node * EMB + o] = bfbits(acc);
    }
}

// ---------------------------------------------------------------------------
// WMMA GEMM + bias + ReLU + residual:
//   out[n, :] = relu(m[n, :] @ W.T + b) + res[n, :]
// A-panel (16 nodes x 256 K) is converted f32->bf16 ONCE per block and staged
// in LDS (padded stride); K-loop is pure ds_load + global b128 + wmma.
// B operand (weights) pre-converted to bf16 -> v16bf (2x b128) loads.
// Block = 256 threads (8 waves); wave w computes two 16x16 output tiles.
// ---------------------------------------------------------------------------
__global__ void __launch_bounds__(256)
gcn_gemm_wmma(const float* __restrict__ m, const __bf16* __restrict__ Wbf,
              const float* __restrict__ bias, const float* __restrict__ res,
              float* __restrict__ out, unsigned short* __restrict__ outbf) {
    __shared__ __bf16 sA[16 * A_STRIDE];

    const int tid = threadIdx.x;
    const int nodeBase = blockIdx.x * 16;

    // ---- stage A: 16 rows x 256 K, one conversion per element per block ----
    {
        int r  = tid >> 4;            // 0..15
        int k0 = (tid & 15) * 16;     // 0..240
        const float* srcp = m + (size_t)(nodeBase + r) * EMB + k0;
        __bf16* dstp = sA + r * A_STRIDE + k0;
#pragma unroll
        for (int i = 0; i < 16; i += 8) {
            v8f f = *(const v8f*)(srcp + i);
#pragma unroll
            for (int j = 0; j < 8; ++j) dstp[i + j] = f2bf(f[j]);
        }
    }
    __syncthreads();

    const int wave = tid >> 5;
    const int lane = tid & 31;
    const int row  = lane & 15;     // M index for A, N index for B/D
    const int hi   = lane >> 4;     // lane half selects K sub-block
    const int cb0  = wave * 32;
    const int cb1  = cb0 + 16;

    v8f acc0 = {}; v8f acc1 = {};
    const __bf16* arow  = sA + row * A_STRIDE + hi * 16;
    const __bf16* wrow0 = Wbf + (size_t)(cb0 + row) * EMB + hi * 16;
    const __bf16* wrow1 = Wbf + (size_t)(cb1 + row) * EMB + hi * 16;

    for (int kb = 0; kb < EMB; kb += 32) {
        // A fragment from LDS: K = kb+hi*16+0..15 (contiguous)
        v8bf a0 = *(const v8bf*)(arow + kb);
        v8bf a1 = *(const v8bf*)(arow + kb + 8);
        v16bf a = __builtin_shufflevector(a0, a1,
                    0, 1, 2, 3, 4, 5, 6, 7, 8, 9, 10, 11, 12, 13, 14, 15);
        // B tiles: 16 contiguous bf16 per lane (32B aligned)
        v16bf b0 = *(const v16bf*)(wrow0 + kb);
        v16bf b1 = *(const v16bf*)(wrow1 + kb);
        acc0 = __builtin_amdgcn_wmma_f32_16x16x32_bf16(
            false, a, false, b0, (short)0, acc0, false, false);
        acc1 = __builtin_amdgcn_wmma_f32_16x16x32_bf16(
            false, a, false, b1, (short)0, acc1, false, false);
    }

    // D layout: VGPR r -> M = r + hi*8, N = row
    float bb0 = bias[cb0 + row];
    float bb1 = bias[cb1 + row];
#pragma unroll
    for (int r = 0; r < 8; ++r) {
        int node = nodeBase + r + hi * 8;
        size_t o0 = (size_t)node * EMB + cb0 + row;
        size_t o1 = (size_t)node * EMB + cb1 + row;
        float v0 = acc0[r] + bb0;
        float v1 = acc1[r] + bb1;
        v0 = v0 > 0.0f ? v0 : 0.0f;
        v1 = v1 > 0.0f ? v1 : 0.0f;
        v0 += res[o0];
        v1 += res[o1];
        out[o0] = v0;
        out[o1] = v1;
        if (outbf) {
            outbf[o0] = bfbits(v0);
            outbf[o1] = bfbits(v1);
        }
    }
}

// ---------------------------------------------------------------------------
// Graph sum-pool: g[seg[n]] += h[n]   (block = node, thread = feature)
// ---------------------------------------------------------------------------
__global__ void __launch_bounds__(256)
pool_kernel(const float* __restrict__ h, const int* __restrict__ seg,
            float* __restrict__ g) {
    int n = blockIdx.x;
    int c = threadIdx.x;
    atomicAdd(&g[(size_t)seg[n] * EMB + c], h[(size_t)n * EMB + c]);
}

// ---------------------------------------------------------------------------
// Pred head: relu(g @ Wp1.T) @ Wp2.T + bp2   (block = graph, 128 threads)
// ---------------------------------------------------------------------------
__global__ void __launch_bounds__(128)
head_kernel(const float* __restrict__ g, const float* __restrict__ Wp1,
            const float* __restrict__ Wp2, const float* __restrict__ bp2,
            float* __restrict__ outp) {
    __shared__ float sh[HID];
    int b = blockIdx.x, j = threadIdx.x;
    const float* gr = g + (size_t)b * EMB;
    const float* wr = Wp1 + (size_t)j * EMB;
    float acc = 0.0f;
#pragma unroll 8
    for (int k = 0; k < EMB; ++k) acc += gr[k] * wr[k];
    acc = acc > 0.0f ? acc : 0.0f;
    sh[j] = acc * Wp2[j];
    __syncthreads();
    for (int off = HID / 2; off > 0; off >>= 1) {
        if (j < off) sh[j] += sh[j + off];
        __syncthreads();
    }
    if (j == 0) outp[b] = sh[0] + bp2[0];
}

// ---------------------------------------------------------------------------
extern "C" void kernel_launch(void* const* d_in, const int* in_sizes, int n_in,
                              void* d_out, int out_size, void* d_ws, size_t ws_size,
                              hipStream_t stream) {
    (void)in_sizes; (void)n_in; (void)out_size; (void)ws_size;
    const float* x     = (const float*)d_in[0];
    const int*   src   = (const int*)d_in[1];
    const int*   dst   = (const int*)d_in[2];
    const int*   seg   = (const int*)d_in[3];
    const float* W_emb = (const float*)d_in[4];
    const float* W1    = (const float*)d_in[5];
    const float* b1    = (const float*)d_in[6];
    const float* W2    = (const float*)d_in[7];
    const float* b2    = (const float*)d_in[8];
    const float* W3    = (const float*)d_in[9];
    const float* b3    = (const float*)d_in[10];
    const float* Wp1   = (const float*)d_in[11];
    const float* Wp2   = (const float*)d_in[12];
    const float* bp2   = (const float*)d_in[13];
    float* out = (float*)d_out;

    const size_t NF = (size_t)N_NODES * EMB;
    char* ws = (char*)d_ws;
    float*          h1   = (float*)ws;          ws += NF * sizeof(float);  // later h3
    float*          h2   = (float*)ws;          ws += NF * sizeof(float);
    float*          mm   = (float*)ws;          ws += NF * sizeof(float);  // m8 & m256
    unsigned short* h1bf = (unsigned short*)ws; ws += NF * sizeof(unsigned short);
    unsigned short* h2bf = (unsigned short*)ws; ws += NF * sizeof(unsigned short);
    float*          h0   = (float*)ws;          ws += (size_t)N_NODES * IN_CH * sizeof(float);
    float*          g    = (float*)ws;          ws += (size_t)N_GRAPHS * EMB * sizeof(float);
    unsigned short* w2bf = (unsigned short*)ws; ws += (size_t)EMB * EMB * sizeof(unsigned short);
    unsigned short* w3bf = (unsigned short*)ws; ws += (size_t)EMB * EMB * sizeof(unsigned short);

    // ---- one-time weight conversion (tiny) ----
    cvtw_kernel<<<(EMB * EMB + 255) / 256, 256, 0, stream>>>(W2, w2bf, EMB * EMB);
    cvtw_kernel<<<(EMB * EMB + 255) / 256, 256, 0, stream>>>(W3, w3bf, EMB * EMB);

    // ---- embedding + layer 1 (K=8) ----
    emb_kernel<<<(N_NODES * IN_CH + 255) / 256, 256, 0, stream>>>(x, W_emb, h0);
    zero_kernel<<<2048, 256, 0, stream>>>(mm, (size_t)N_NODES * IN_CH);
    scatter8_kernel<<<(N_EDGES + 255) / 256, 256, 0, stream>>>(h0, src, dst, mm);
    layer1_kernel<<<N_NODES / 16, 256, 0, stream>>>(mm, W1, b1, h1, h1bf);

    // ---- layer 2 (WMMA, residual) ----
    zero_kernel<<<8192, 256, 0, stream>>>(mm, NF);
    scatter256_kernel<<<N_EDGES, 256, 0, stream>>>(h1bf, src, dst, mm);
    gcn_gemm_wmma<<<N_NODES / 16, 256, 0, stream>>>(mm, (const __bf16*)w2bf, b2, h1, h2, h2bf);

    // ---- layer 3 (WMMA, residual); f32 result into h1 buffer ----
    zero_kernel<<<8192, 256, 0, stream>>>(mm, NF);
    scatter256_kernel<<<N_EDGES, 256, 0, stream>>>(h2bf, src, dst, mm);
    gcn_gemm_wmma<<<N_NODES / 16, 256, 0, stream>>>(mm, (const __bf16*)w3bf, b3, h2, h1, (unsigned short*)nullptr);

    // ---- pooling + head ----
    zero_kernel<<<1024, 256, 0, stream>>>(g, (size_t)N_GRAPHS * EMB);
    pool_kernel<<<N_NODES, 256, 0, stream>>>(h1, seg, g);
    head_kernel<<<N_GRAPHS, HID, 0, stream>>>(g, Wp1, Wp2, bp2, out);
}
